// mmd_rbf_loss_9242769621335
// MI455X (gfx1250) — matrix operand order repaired and checked by reading, not compiled, not run on hardware
//
#include <hip/hip_runtime.h>

typedef __attribute__((ext_vector_type(16))) __bf16          v16bf;
typedef __attribute__((ext_vector_type(8)))  float           v8f;
typedef __attribute__((ext_vector_type(16))) unsigned short  ushort16;
typedef __attribute__((ext_vector_type(4)))  int             v4i;

#define N_TOTAL 8192
#define NHALF   4096
#define DCOL    256
#define NBT     128                       // 8192/64 tiles per side
#define NTILES  (NBT * (NBT + 1) / 2)     // 8256 triangular tiles

// workspace layout (float offsets)
#define WS_SQ      0
#define WS_CSPART  8192                    // 64*256 colsum partials
#define WS_INVBW   (8192 + 16384)          // 5 floats
#define WS_TILES   (8192 + 16384 + 8)      // NTILES tile partials

#if defined(__has_builtin)
#if __has_builtin(__builtin_amdgcn_global_load_async_to_lds_b128)
#define HAVE_ASYNC_COPY 1
#endif
#endif

#ifdef HAVE_ASYNC_COPY
typedef __attribute__((address_space(1))) v4i gv4i;   // global int4
typedef __attribute__((address_space(3))) v4i lv4i;   // LDS int4
__device__ __forceinline__ void async_copy16(const float* g, void* l) {
  __builtin_amdgcn_global_load_async_to_lds_b128((gv4i*)g, (lv4i*)l, 0, 0);
}
__device__ __forceinline__ void wait_async0() {
#if __has_builtin(__builtin_amdgcn_s_wait_asynccnt)
  __builtin_amdgcn_s_wait_asynccnt(0);
#else
  asm volatile("s_wait_asynccnt 0x0" ::: "memory");
#endif
}
#endif

__device__ __forceinline__ const float* row_ptr(const float* __restrict__ src,
                                                const float* __restrict__ tgt, int r) {
  return (r < NHALF) ? (src + (size_t)r * DCOL) : (tgt + (size_t)(r - NHALF) * DCOL);
}

__device__ __forceinline__ unsigned short f32_bf16_rne(float f) {
  unsigned u = __float_as_uint(f);
  unsigned r = 0x7FFFu + ((u >> 16) & 1u);
  return (unsigned short)((u + r) >> 16);
}
__device__ __forceinline__ float bf16_f32(unsigned short h) {
  return __uint_as_float(((unsigned)h) << 16);
}

__device__ __forceinline__ v8f wmma_bf16(v16bf a, v16bf b, v8f c) {
  // v_wmma_f32_16x16x32_bf16: (neg_a, A, neg_b, B, c_mod, C, reuse_a, reuse_b)
  return __builtin_amdgcn_wmma_f32_16x16x32_bf16(false, a, false, b, (short)0, c, false, false);
}

// ---------------- row norms: one wave32 per row ----------------
__global__ void rownorm_kernel(const float* __restrict__ src, const float* __restrict__ tgt,
                               float* __restrict__ sq) {
  int wave = threadIdx.x >> 5, lane = threadIdx.x & 31;
  int r = blockIdx.x * 8 + wave;
  const float* p = row_ptr(src, tgt, r);
  float s = 0.f;
#pragma unroll
  for (int e = 0; e < 8; ++e) { float v = p[lane + 32 * e]; s = fmaf(v, v, s); }
#pragma unroll
  for (int off = 16; off > 0; off >>= 1) s += __shfl_down(s, off, 32);
  if (lane == 0) sq[r] = s;
}

// ---------------- column-sum partials (deterministic) ----------------
__global__ void colsum_kernel(const float* __restrict__ src, const float* __restrict__ tgt,
                              float* __restrict__ csPart) {
  int t = threadIdx.x, b = blockIdx.x;
  float s = 0.f;
  for (int r = b * 128; r < b * 128 + 128; ++r) s += row_ptr(src, tgt, r)[t];
  csPart[b * 256 + t] = s;
}

// ---------------- bandwidth: sum_l2 = 2N*sum(sq) - 2*||colsum||^2 ----------------
__global__ void bandwidth_kernel(const float* __restrict__ sq, const float* __restrict__ csPart,
                                 float* __restrict__ invbw) {
  __shared__ float red[256];
  int t = threadIdx.x;
  float s = 0.f;
  for (int k = 0; k < 32; ++k) s += sq[t + 256 * k];
  red[t] = s; __syncthreads();
  for (int off = 128; off > 0; off >>= 1) { if (t < off) red[t] += red[t + off]; __syncthreads(); }
  float sumsq = red[0]; __syncthreads();

  float cs = 0.f;
  for (int b = 0; b < 64; ++b) cs += csPart[b * 256 + t];
  red[t] = cs * cs; __syncthreads();
  for (int off = 128; off > 0; off >>= 1) { if (t < off) red[t] += red[t + off]; __syncthreads(); }

  if (t == 0) {
    float cs2 = red[0];
    float sum_l2 = 2.f * (float)N_TOTAL * sumsq - 2.f * cs2;
    float bw = sum_l2 / ((float)N_TOTAL * (float)N_TOTAL - (float)N_TOTAL);
    bw = bw * 0.25f;                       // / KERNEL_MUL^(KERNEL_NUM//2) = /4
    float m = 1.f;
#pragma unroll
    for (int q = 0; q < 5; ++q) { invbw[q] = 1.f / (bw * m); m *= 2.f; }
  }
}

// ---------------- main fused tile kernel: 64x64 output tile per block ----------------
#define LDK 72   // LDS row stride in ushorts (padding vs bank conflicts)

__global__ void __launch_bounds__(128) mmd_tile_kernel(const float* __restrict__ src,
                                                       const float* __restrict__ tgt,
                                                       const float* __restrict__ sq,
                                                       const float* __restrict__ invbw,
                                                       float* __restrict__ tilePart) {
  __shared__ unsigned short Ahi[64][LDK], Alo[64][LDK], Bhi[64][LDK], Blo[64][LDK];
#ifdef HAVE_ASYNC_COPY
  __shared__ float Af32[64][64], Bf32[64][64];
#endif
  __shared__ float sqA[64], sqB[64], red[128];

  int tid = threadIdx.x;
  int lane = tid & 31, wave = tid >> 5;

  // triangular tile decode: tile t -> (bi, bj) with bj <= bi
  int t = blockIdx.x;
  int bi = (int)((sqrtf(8.f * (float)t + 1.f) - 1.f) * 0.5f);
  while ((bi + 1) * (bi + 2) / 2 <= t) ++bi;
  while (bi * (bi + 1) / 2 > t) --bi;
  int bj = t - bi * (bi + 1) / 2;
  int i0 = bi * 64, j0 = bj * 64;

  if (tid < 64) sqA[tid] = sq[i0 + tid];
  else          sqB[tid - 64] = sq[j0 + (tid - 64)];

  float ib[5];
#pragma unroll
  for (int q = 0; q < 5; ++q) ib[q] = invbw[q];

  v8f zero = {0.f, 0.f, 0.f, 0.f, 0.f, 0.f, 0.f, 0.f};
  v8f acc[4];
#pragma unroll
  for (int mt = 0; mt < 4; ++mt) acc[mt] = zero;

  for (int kc0 = 0; kc0 < 256; kc0 += 64) {
    __syncthreads();
#ifdef HAVE_ASYNC_COPY
    // async global -> LDS copy of raw fp32 chunks (ASYNCcnt path)
    for (int idx = tid; idx < 64 * 16; idx += 128) {
      int r = idx >> 4, c4 = (idx & 15) << 2;
      async_copy16(row_ptr(src, tgt, i0 + r) + kc0 + c4, &Af32[r][c4]);
      async_copy16(row_ptr(src, tgt, j0 + r) + kc0 + c4, &Bf32[r][c4]);
    }
    wait_async0();
    __syncthreads();
    // LDS fp32 -> bf16 hi/lo split
    for (int idx = tid; idx < 64 * 64; idx += 128) {
      int r = idx >> 6, c = idx & 63;
      float a = Af32[r][c];
      unsigned short h = f32_bf16_rne(a);
      Ahi[r][c] = h;
      Alo[r][c] = f32_bf16_rne(a - bf16_f32(h));
      float b = Bf32[r][c];
      h = f32_bf16_rne(b);
      Bhi[r][c] = h;
      Blo[r][c] = f32_bf16_rne(b - bf16_f32(h));
    }
#else
    // fallback: stage through VGPRs with inline bf16 split
    for (int idx = tid; idx < 64 * 16; idx += 128) {
      int r = idx >> 4, c4 = (idx & 15) << 2;
      const float4 va = *reinterpret_cast<const float4*>(row_ptr(src, tgt, i0 + r) + kc0 + c4);
      const float4 vb = *reinterpret_cast<const float4*>(row_ptr(src, tgt, j0 + r) + kc0 + c4);
      float aa[4] = {va.x, va.y, va.z, va.w};
      float bb[4] = {vb.x, vb.y, vb.z, vb.w};
#pragma unroll
      for (int c = 0; c < 4; ++c) {
        unsigned short h = f32_bf16_rne(aa[c]);
        Ahi[r][c4 + c] = h;
        Alo[r][c4 + c] = f32_bf16_rne(aa[c] - bf16_f32(h));
        h = f32_bf16_rne(bb[c]);
        Bhi[r][c4 + c] = h;
        Blo[r][c4 + c] = f32_bf16_rne(bb[c] - bf16_f32(h));
      }
    }
#endif
    __syncthreads();

#pragma unroll
    for (int kc = 0; kc < 64; kc += 32) {
      // B fragment: lane L holds column N = L%16; K = {0..7,16..23} (+8 for lanes>=16) + kc
      int nrow = (wave << 4) + (lane & 15);          // row of X_B == column of B
      int kbB  = kc + ((lane >> 4) << 3);
      ushort16 ubh, ubl;
#pragma unroll
      for (int e = 0; e < 16; ++e) { int kb = kbB + (e & 7) + ((e >> 3) << 4); ubh[e] = Bhi[nrow][kb]; ubl[e] = Blo[nrow][kb]; }
      v16bf bh = __builtin_bit_cast(v16bf, ubh);
      v16bf bl = __builtin_bit_cast(v16bf, ubl);

#pragma unroll
      for (int mt = 0; mt < 4; ++mt) {
        // A fragment: lane L holds row M = L%16; K = {0..7,16..23} (+8 for lanes>=16)
        int mrow = (mt << 4) + (lane & 15);
        int kbA  = kc + ((lane >> 4) << 3);
        ushort16 uah, ual;
#pragma unroll
        for (int e = 0; e < 16; ++e) {
          int k = kbA + (e & 7) + ((e >> 3) << 4);
          uah[e] = Ahi[mrow][k]; ual[e] = Alo[mrow][k];
        }
        v16bf ahv = __builtin_bit_cast(v16bf, uah);
        v16bf alv = __builtin_bit_cast(v16bf, ual);
        // fp32 emulation via bf16 split: hi*hi + hi*lo + lo*hi
        acc[mt] = wmma_bf16(ahv, bh, acc[mt]);
        acc[mt] = wmma_bf16(ahv, bl, acc[mt]);
        acc[mt] = wmma_bf16(alv, bh, acc[mt]);
      }
    }
  }

  // epilogue: l2 -> 5-scale RBF -> signed reduction
  float part = 0.f;
  float sqj = sqB[(wave << 4) + (lane & 15)];
  int moff = (lane >> 4) << 3;                        // C/D layout: M = v + 8*(lane>=16)
#pragma unroll
  for (int mt = 0; mt < 4; ++mt) {
#pragma unroll
    for (int v = 0; v < 8; ++v) {
      float g = acc[mt][v];
      float sqi = sqA[(mt << 4) + moff + v];
      float l2 = fmaxf(sqi + sqj - 2.f * g, 0.f);
#pragma unroll
      for (int q = 0; q < 5; ++q) part += __expf(-l2 * ib[q]);
    }
  }
  red[tid] = part; __syncthreads();
  for (int off = 64; off > 0; off >>= 1) { if (tid < off) red[tid] += red[tid + off]; __syncthreads(); }
  if (tid == 0) {
    float sign = ((i0 < NHALF) == (j0 < NHALF)) ? 1.f : -1.f;
    float weight = (bi == bj) ? 1.f : 2.f;            // symmetric matrix: count lower tiles twice
    tilePart[blockIdx.x] = sign * weight * red[0];
  }
}

// ---------------- deterministic final reduction ----------------
__global__ void finalize_kernel(const float* __restrict__ tilePart, float* __restrict__ out) {
  __shared__ float red[256];
  int t = threadIdx.x;
  float s = 0.f;
  for (int k = t; k < NTILES; k += 256) s += tilePart[k];
  red[t] = s; __syncthreads();
  for (int off = 128; off > 0; off >>= 1) { if (t < off) red[t] += red[t + off]; __syncthreads(); }
  if (t == 0) out[0] = red[0] / ((float)NHALF * (float)NHALF);
}

extern "C" void kernel_launch(void* const* d_in, const int* in_sizes, int n_in,
                              void* d_out, int out_size, void* d_ws, size_t ws_size,
                              hipStream_t stream) {
  const float* src = (const float*)d_in[0];
  const float* tgt = (const float*)d_in[1];
  float* ws       = (float*)d_ws;
  float* sq       = ws + WS_SQ;
  float* csPart   = ws + WS_CSPART;
  float* invbw    = ws + WS_INVBW;
  float* tilePart = ws + WS_TILES;
  float* out      = (float*)d_out;

  rownorm_kernel  <<<1024,  256, 0, stream>>>(src, tgt, sq);
  colsum_kernel   <<<64,    256, 0, stream>>>(src, tgt, csPart);
  bandwidth_kernel<<<1,     256, 0, stream>>>(sq, csPart, invbw);
  mmd_tile_kernel <<<NTILES,128, 0, stream>>>(src, tgt, sq, invbw, tilePart);
  finalize_kernel <<<1,     256, 0, stream>>>(tilePart, out);
}
